// gen_C_11347303596498
// MI455X (gfx1250) — matrix operand order repaired and checked by reading, not compiled, not run on hardware
//
#include <hip/hip_runtime.h>

typedef __attribute__((ext_vector_type(16))) _Float16 v16h;
typedef __attribute__((ext_vector_type(8)))  _Float16 v8h;
typedef __attribute__((ext_vector_type(8)))  float    v8f;

// Problem constants (from reference)
#define MROWS   132000          // N * 2 * (KNN+1)
#define DDIM    12000           // MODES * N
#define NODES   64
#define NLAYERS 3               // LC - 1 hidden layers
#define ROWS    64              // rows per workgroup
#define STR     72              // padded f16 row stride (multiple of 8 -> 16B aligned ds_load_b128)

__global__ void zero_out_kernel(float* __restrict__ out, long long n) {
    long long i = (long long)blockIdx.x * blockDim.x + threadIdx.x;
    long long stride = (long long)gridDim.x * blockDim.x;
    long long n4 = n >> 2;
    float4* o4 = reinterpret_cast<float4*>(out);
    float4 z; z.x = 0.f; z.y = 0.f; z.z = 0.f; z.w = 0.f;
    for (long long j = i; j < n4; j += stride) o4[j] = z;
    for (long long j = (n4 << 2) + i; j < n; j += stride) out[j] = 0.f;
}

__global__ __launch_bounds__(256) void mlp_wmma_scatter_kernel(
    const float* __restrict__ CK,    // [M,3]
    const float* __restrict__ Win,   // [3,64]
    const float* __restrict__ bin,   // [64]
    const float* __restrict__ Whid,  // [3,64,64]
    const float* __restrict__ bhid,  // [3,64]
    const float* __restrict__ Wout,  // [64,4]
    const float* __restrict__ bout,  // [4]
    const int*   __restrict__ coo,   // [2,M]
    float* __restrict__ out)         // [D,D]
{
    __shared__ _Float16 sH0[ROWS * STR];            // activations ping
    __shared__ _Float16 sH1[ROWS * STR];            // activations pong
    __shared__ _Float16 sWt[NLAYERS * NODES * STR]; // transposed hidden weights: Wt[l][n][k]
    __shared__ float sWin[3 * NODES];
    __shared__ float sBin[NODES];
    __shared__ float sBhid[NLAYERS * NODES];
    __shared__ float sWout[NODES * 4];
    __shared__ float sBout[4];
    __shared__ float sX[ROWS * 3];
    __shared__ float sC[ROWS * 4];

    const int t    = threadIdx.x;
    const int wave = t >> 5;
    const int lane = t & 31;
    const int m16  = lane & 15;
    const int h    = lane >> 4;
    const int r0   = blockIdx.x * ROWS;   // first global row of this block

    // ---- Stage weights / biases / inputs into LDS ----
    for (int idx = t; idx < 3 * NODES; idx += 256) sWin[idx] = Win[idx];
    for (int idx = t; idx < NODES; idx += 256)     sBin[idx] = bin[idx];
    for (int idx = t; idx < NLAYERS * NODES * NODES; idx += 256) {
        int l = idx >> 12;            // /4096
        int r = idx & 4095;
        int k = r >> 6;
        int n = r & 63;
        sWt[(l * NODES + n) * STR + k] = (_Float16)Whid[idx];
    }
    for (int idx = t; idx < NLAYERS * NODES; idx += 256) sBhid[idx] = bhid[idx];
    for (int idx = t; idx < NODES * 4; idx += 256)       sWout[idx] = Wout[idx];
    if (t < 4) sBout[t] = bout[t];
    for (int idx = t; idx < ROWS * 3; idx += 256) {
        int g = r0 * 3 + idx;
        sX[idx] = (g < MROWS * 3) ? CK[g] : 0.0f;
    }
    __syncthreads();

    // ---- Layer 0 (3 -> 64), scalar: K too small for WMMA ----
    for (int idx = t; idx < ROWS * NODES; idx += 256) {
        int row = idx >> 6;
        int col = idx & 63;
        float s = sBin[col]
                + sX[row * 3 + 0] * sWin[0 * NODES + col]
                + sX[row * 3 + 1] * sWin[1 * NODES + col]
                + sX[row * 3 + 2] * sWin[2 * NODES + col];
        s = fmaxf(s, 0.0f);
        sH0[row * STR + col] = (_Float16)s;
    }
    __syncthreads();

    // ---- Hidden layers (64 -> 64) via v_wmma_f32_16x16x32_f16 ----
    _Float16* cur = sH0;
    _Float16* nxt = sH1;
    for (int layer = 0; layer < NLAYERS; ++layer) {
        const _Float16* Wt = &sWt[layer * NODES * STR];
        const float* bias  = &sBhid[layer * NODES];

        #pragma unroll
        for (int tt = 0; tt < 2; ++tt) {
            int tile = wave + tt * 8;     // 16 tiles of 16x16, 2 per wave
            int rb = tile >> 2;           // row block 0..3
            int cb = tile & 3;            // col block 0..3

            v8f acc = {};
            #pragma unroll
            for (int ks = 0; ks < 2; ++ks) {
                // A fragment: 16-bit A 16x32 layout; row = rb*16 + (lane&15)
                // elements 0..7 : K = ks*32 + 8h + j ; elements 8..15 : K = ks*32 + 16 + 8h + j
                const _Float16* ap = cur + (rb * 16 + m16) * STR + ks * 32 + 8 * h;
                v8h aLo = *reinterpret_cast<const v8h*>(ap);
                v8h aHi = *reinterpret_cast<const v8h*>(ap + 16);
                v16h a;
                #pragma unroll
                for (int j = 0; j < 8; ++j) { a[j] = aLo[j]; a[j + 8] = aHi[j]; }

                // B fragment: 16-bit B 32x16 layout; col n = cb*16 + (lane&15)
                // elements 0..7 : K = ks*32 + 8h + j ; elements 8..15 : K = ks*32 + 16 + 8h + j  (Wt stored as [n][k] so reads are contiguous)
                const _Float16* bp = Wt + (cb * 16 + m16) * STR + ks * 32 + 8 * h;
                v8h bLo = *reinterpret_cast<const v8h*>(bp);
                v8h bHi = *reinterpret_cast<const v8h*>(bp + 16);
                v16h b;
                #pragma unroll
                for (int j = 0; j < 8; ++j) { b[j] = bLo[j]; b[j + 8] = bHi[j]; }

                acc = __builtin_amdgcn_wmma_f32_16x16x32_f16(
                          false, a, false, b, (short)0, acc, false, false);
            }

            // D layout: element r -> row = rb*16 + r + 8h, col = cb*16 + (lane&15)
            float bcol = bias[cb * 16 + m16];
            #pragma unroll
            for (int r = 0; r < 8; ++r) {
                float v = acc[r] + bcol;
                v = fmaxf(v, 0.0f);
                nxt[(rb * 16 + r + 8 * h) * STR + cb * 16 + m16] = (_Float16)v;
            }
        }
        __syncthreads();
        _Float16* tmp = cur; cur = nxt; nxt = tmp;
    }

    // ---- Output layer (64 -> 4), scalar: 256 threads = 64 rows x 4 channels ----
    {
        int row = t >> 2;
        int ch  = t & 3;
        float s = sBout[ch];
        #pragma unroll
        for (int k = 0; k < NODES; ++k)
            s += (float)cur[row * STR + k] * sWout[k * 4 + ch];
        sC[row * 4 + ch] = s;
    }
    __syncthreads();

    // ---- Channel-pair sum + scatter-add: vals[mj] = C[mj] + C[2+mj] ----
    if (t < ROWS * 2) {
        int row = t >> 1;
        int mj  = t & 1;
        int gm  = r0 + row;
        if (gm < MROWS) {
            float val = sC[row * 4 + mj] + sC[row * 4 + 2 + mj];
            int c0 = coo[gm];
            int c1 = coo[MROWS + gm];
            long long idx = (long long)(c0 * 2 + mj) * DDIM + (c1 * 2 + mj);
            atomicAdd(&out[idx], val);
        }
    }
}

extern "C" void kernel_launch(void* const* d_in, const int* in_sizes, int n_in,
                              void* d_out, int out_size, void* d_ws, size_t ws_size,
                              hipStream_t stream) {
    const float* CK   = (const float*)d_in[0];
    const float* Win  = (const float*)d_in[1];
    const float* bin  = (const float*)d_in[2];
    const float* Whid = (const float*)d_in[3];
    const float* bhid = (const float*)d_in[4];
    const float* Wout = (const float*)d_in[5];
    const float* bout = (const float*)d_in[6];
    const int*   coo  = (const int*)d_in[7];
    float* out = (float*)d_out;

    // 576 MB zero-fill: the bandwidth-dominant step (~25 us at 23.3 TB/s)
    zero_out_kernel<<<4096, 256, 0, stream>>>(out, (long long)out_size);

    int blocks = (MROWS + ROWS - 1) / ROWS;
    mlp_wmma_scatter_kernel<<<blocks, 256, 0, stream>>>(
        CK, Win, bin, Whid, bhid, Wout, bout, coo, out);
}
